// PocketSE3Transformer_58523224375464
// MI455X (gfx1250) — matrix operand order repaired and checked by reading, not compiled, not run on hardware
//
#include <hip/hip_runtime.h>
#include <hip/hip_bf16.h>

typedef __bf16 bf16;
typedef __attribute__((ext_vector_type(16))) __bf16 v16bf;
typedef __attribute__((ext_vector_type(8)))  float  v8f;

#define HS 32
#define HV 16
#define NH 8
#define NB 32
#define KTOT 129            // 128 hidden channels + 1 bias channel
#define RSTRIDE 64          // padded column stride of repacked R (48 used)
#define INV_SQRT_HS 0.17677669529663687f
#define INV_SQRT_HV 0.25f
#define INV_SQRT_NH 0.35355339059327373f
#define SQ3F 1.7320508075688772f
#define PI_F 3.14159265358979f

// fast transcendentals: v_rcp_f32 / v_exp_f32 are TRANS ops that co-issue
// with the WMMA (XDL) pipe -- avoid IEEE div expansion (div_scale/div_fixup).
__device__ __forceinline__ float fast_rcp(float x){ return __builtin_amdgcn_rcpf(x); }
__device__ __forceinline__ float silu_f(float x){ return x * fast_rcp(1.0f + __expf(-x)); }
__device__ __forceinline__ float sigmoid_f(float x){ return fast_rcp(1.0f + __expf(-x)); }

static __device__ __forceinline__ v8f wmma_bf16(v16bf a, v16bf b, v8f c){
  // D = A(16x32 bf16) * B(32x16 bf16) + C(16x16 f32)
  return __builtin_amdgcn_wmma_f32_16x16x32_bf16(false, a, false, b, (short)0, c, false, false);
}

static __device__ __forceinline__ v16bf splat16(bf16 x){
  v16bf r;
  #pragma unroll
  for (int i=0;i<16;i++) r[i] = x;
  return r;
}

// ---------------------------------------------------------------- utilities
__global__ void zero_kernel(float* p, long n){
  long i = (long)blockIdx.x*blockDim.x + threadIdx.x;
  if (i < n) p[i] = 0.0f;
}

__global__ void prep_r1_kernel(const float* r1W, bf16* r1Wb){
  int idx = blockIdx.x*blockDim.x + threadIdx.x;
  if (idx < HS*128) r1Wb[idx] = (bf16)r1W[idx];
}

// Repack r2_W's ss (cols [0,1024)) and sv (cols [1536,2048)) blocks into
// R[(kk*32+i)*64 + j], j<32 -> msg_s col j, 32<=j<48 -> a_sv col (j-32).
// kk==128 is the bias channel (r2_b).
__global__ void prep_R_kernel(const float* r2W, const float* r2b, bf16* R){
  int idx = blockIdx.x*blockDim.x + threadIdx.x;
  int total = KTOT*HS*RSTRIDE;
  if (idx >= total) return;
  int col = idx % RSTRIDE;
  int row = idx / RSTRIDE;
  int kk  = row / HS;
  int i   = row % HS;
  float v = 0.0f;
  if (col < HS){
    int sc = i*HS + col;
    v = (kk < 128) ? r2W[(size_t)kk*2304 + sc] : r2b[sc];
  } else if (col < HS+HV){
    int sc = 1536 + i*HV + (col-HS);
    v = (kk < 128) ? r2W[(size_t)kk*2304 + sc] : r2b[sc];
  }
  R[idx] = (bf16)v;
}

// x_s = features @ embed_W + b  (f32 + bf16 copies), node_q = x_s @ q_W + q_b
__global__ void embed_kernel(const float* feat, const float* eW, const float* eb,
                             const float* qW, const float* qb,
                             float* xsf, bf16* xsb, float* nodeq, int N){
  int i = blockIdx.x*blockDim.x + threadIdx.x;
  if (i >= N) return;
  const float* f = feat + (size_t)i*46;
  float xs[HS];
  for (int j=0;j<HS;j++){
    float acc = eb[j];
    for (int k=0;k<46;k++) acc += f[k]*eW[k*HS+j];
    xs[j] = acc;
    xsf[(size_t)i*HS+j] = acc;
    xsb[(size_t)i*HS+j] = (bf16)acc;
  }
  for (int h=0;h<NH;h++){
    float acc = qb[h];
    for (int j=0;j<HS;j++) acc += xs[j]*qW[j*NH+h];
    nodeq[(size_t)i*NH+h] = acc;
  }
}

// -------------------------------------------------------------- edge kernel
// One wave per 32-edge group (two 16-row WMMA tiles) so each streamed
// B-fragment triple feeds 6 WMMAs.  GEMM1: h = silu(rbf @ r1W + b).
// GEMM2: [msg_s | a_sv] = P @ R, A fragments synthesized as h[e,kk]*xs[e,:]
// via packed bf16 multiplies.
#define WPB 2
__global__ void __launch_bounds__(64) edge_kernel(
    const int* __restrict__ ei, const float* __restrict__ pos,
    const bf16* __restrict__ xsb, const float* __restrict__ nodeq,
    const bf16* __restrict__ r1Wb, const float* __restrict__ r1b,
    const bf16* __restrict__ R,
    const float* __restrict__ kW, const float* __restrict__ kb,
    float* __restrict__ msg_s, float* __restrict__ a_sv,
    float* __restrict__ shv, float* __restrict__ logit,
    unsigned int* __restrict__ nmax, int E, int ngroups)
{
  __shared__ bf16  s_rbf[WPB][32][NB];
  __shared__ bf16  s_xs [WPB][32][HS];
  __shared__ bf16  s_h  [WPB][32][132];
  __shared__ float s_out[WPB][32][48];
  __shared__ float s_env[WPB][32];
  __shared__ float s_sh [WPB][32][3];
  __shared__ int   s_dst[WPB][32];

  const int wave  = threadIdx.x >> 5;
  const int lane  = threadIdx.x & 31;
  const int group = blockIdx.x * WPB + wave;
  if (group >= ngroups) return;
  const long e0 = (long)group * 32;

  // ---- per-edge prep: every lane owns one edge of the group ----
  {
    const bool ev = (e0 + lane) < (long)E;
    const long e  = ev ? (e0 + lane) : (long)(E-1);
    const int src = ei[e];
    const int dst = ei[E + e];
    s_dst[wave][lane] = dst;
    float rx = pos[src*3+0]-pos[dst*3+0];
    float ry = pos[src*3+1]-pos[dst*3+1];
    float rz = pos[src*3+2]-pos[dst*3+2];
    float d  = sqrtf(rx*rx+ry*ry+rz*rz + 1e-12f);
    float iv = SQ3F * fast_rcp(d);
    s_sh[wave][lane][0]=rx*iv; s_sh[wave][lane][1]=ry*iv; s_sh[wave][lane][2]=rz*iv;
    s_env[wave][lane] = fmaxf(0.5f*(1.0f + __cosf(PI_F*d*0.2f)), 0.0f);
    const float width = 19.22f;   // 0.5 / (5/31)^2
    #pragma unroll
    for (int g=0; g<NB; g++){
      float t = d - (5.0f/31.0f)*g;
      s_rbf[wave][lane][g] = (bf16)__expf(-width*t*t);
    }
    const bf16* xr = xsb + (size_t)src*HS;
    #pragma unroll
    for (int j=0;j<HS;j++) s_xs[wave][lane][j] = xr[j];
    s_h[wave][lane][128] = (bf16)1.0f;   // bias channel for GEMM2
  }

  const int m     = lane & 15;
  const int n     = lane & 15;
  const int kbase = (lane < 16) ? 0 : 8;
  const int mrow  = (lane >> 4) << 3;

  int kidx[16];                          // 16-bit A-matrix 16x32 layout
  #pragma unroll
  for (int v=0; v<8; v++){
    int k0 = (v<4) ? (kbase + 2*v) : (16 + kbase + 2*(v-4));
    kidx[2*v] = k0; kidx[2*v+1] = k0+1;
  }

  // ---- GEMM1: h = silu(rbf @ r1W + r1b), two 16-edge tiles share B ----
  v16bf a1_0, a1_1;
  #pragma unroll
  for (int q2=0;q2<16;q2++){
    a1_0[q2] = s_rbf[wave][m   ][kidx[q2]];
    a1_1[q2] = s_rbf[wave][16+m][kidx[q2]];
  }
  #pragma unroll
  for (int t=0; t<8; t++){
    v16bf b;                                             // K=kidx (table order), N=n
    #pragma unroll
    for (int q2=0;q2<16;q2++) b[q2] = r1Wb[kidx[q2]*128 + t*16 + n];
    v8f c0t = {}; c0t = wmma_bf16(a1_0, b, c0t);
    v8f c1t = {}; c1t = wmma_bf16(a1_1, b, c1t);
    float bias = r1b[t*16 + n];
    #pragma unroll
    for (int r=0;r<8;r++){
      s_h[wave][   mrow + r][t*16 + n] = (bf16)silu_f(c0t[r] + bias);
      s_h[wave][16+mrow + r][t*16 + n] = (bf16)silu_f(c1t[r] + bias);
    }
  }

  // cache both tiles' xs A-fragment slots as bf16 vectors
  v16bf xvb0, xvb1;
  #pragma unroll
  for (int q2=0;q2<16;q2++){
    xvb0[q2] = s_xs[wave][m   ][kidx[q2]];
    xvb1[q2] = s_xs[wave][16+m][kidx[q2]];
  }

  // ---- GEMM2: [msg_s(32) | a_sv(16)] = P @ R, K = 129*32 ----
  v8f acc00={},acc01={},acc02={};   // tile 0
  v8f acc10={},acc11={},acc12={};   // tile 1
  for (int kk=0; kk<KTOT; kk++){
    bf16 h0 = s_h[wave][m   ][kk];
    bf16 h1 = s_h[wave][16+m][kk];
    v16bf a0 = xvb0 * splat16(h0);   // packed bf16 multiplies
    v16bf a1 = xvb1 * splat16(h1);
    const bf16* rp = R + (size_t)(kk*HS + lane) * RSTRIDE;  // K=lane within chunk
    if (kk + 8 < KTOT)
      __builtin_prefetch(rp + (size_t)8*HS*RSTRIDE, 0, 3);
    v16bf b0, b1, b2;
    #pragma unroll
    for (int q2=0;q2<16;q2++){
      const bf16* rq = R + (size_t)(kk*HS + kidx[q2]) * RSTRIDE + n;
      b0[q2] = rq[0]; b1[q2] = rq[16]; b2[q2] = rq[32];
    }
    acc00 = wmma_bf16(a0, b0, acc00);
    acc10 = wmma_bf16(a1, b0, acc10);
    acc01 = wmma_bf16(a0, b1, acc01);
    acc11 = wmma_bf16(a1, b1, acc11);
    acc02 = wmma_bf16(a0, b2, acc02);
    acc12 = wmma_bf16(a1, b2, acc12);
  }

  #pragma unroll
  for (int r=0;r<8;r++){
    s_out[wave][   mrow+r][n]    = acc00[r]*INV_SQRT_HS;
    s_out[wave][   mrow+r][16+n] = acc01[r]*INV_SQRT_HS;
    s_out[wave][   mrow+r][32+n] = acc02[r]*INV_SQRT_HS;
    s_out[wave][16+mrow+r][n]    = acc10[r]*INV_SQRT_HS;
    s_out[wave][16+mrow+r][16+n] = acc11[r]*INV_SQRT_HS;
    s_out[wave][16+mrow+r][32+n] = acc12[r]*INV_SQRT_HS;
  }

  // ---- epilogue: every lane finalizes its own edge ----
  if ((e0 + lane) < (long)E){
    const long e  = e0 + lane;
    const int dst = s_dst[wave][lane];
    float ms[HS];
    #pragma unroll
    for (int j=0;j<HS;j++){ ms[j] = s_out[wave][lane][j]; msg_s[(size_t)e*HS+j] = ms[j]; }
    #pragma unroll
    for (int j=0;j<HV;j++) a_sv[(size_t)e*HV+j] = s_out[wave][lane][32+j];
    shv[(size_t)e*4+0] = s_sh[wave][lane][0];
    shv[(size_t)e*4+1] = s_sh[wave][lane][1];
    shv[(size_t)e*4+2] = s_sh[wave][lane][2];
    const float* q = nodeq + (size_t)dst*NH;
    float acc = 0.0f;
    #pragma unroll
    for (int h=0;h<NH;h++){
      float kv = kb[h];
      for (int j=0;j<HS;j++) kv += ms[j]*kW[j*NH+h];
      acc += q[h]*kv;
    }
    float lg = acc*INV_SQRT_NH + __logf(s_env[wave][lane] + 1e-8f);
    logit[e] = lg;
    unsigned u = __float_as_uint(lg);
    u = (u & 0x80000000u) ? ~u : (u | 0x80000000u);   // order-preserving encode
    atomicMax(nmax + dst, u);
  }
}

// ------------------------------------------------- softmax + weighted scatter
__global__ void softmax_acc_kernel(const int* ei, const float* logit, const unsigned* nmax,
    const float* msg_s, const float* a_sv, const float* shv,
    float* se, float* aggs, float* aggv, int E)
{
  int e = blockIdx.x*blockDim.x + threadIdx.x;
  if (e >= E) return;
  int dst = ei[E + e];
  unsigned u = nmax[dst];
  float mx = (u & 0x80000000u) ? __uint_as_float(u & 0x7fffffffu) : __uint_as_float(~u);
  float ex = __expf(logit[e] - mx);
  atomicAdd(se + dst, ex);
  const float* msp = msg_s + (size_t)e*HS;
  float* ap = aggs + (size_t)dst*HS;
  for (int j=0;j<HS;j++) atomicAdd(ap + j, ex*msp[j]);
  float s0 = shv[(size_t)e*4+0], s1 = shv[(size_t)e*4+1], s2 = shv[(size_t)e*4+2];
  const float* avp = a_sv + (size_t)e*HV;
  float* agv = aggv + (size_t)dst*48;
  for (int j=0;j<HV;j++){
    float a = ex*avp[j];
    atomicAdd(agv + j*3+0, a*s0);
    atomicAdd(agv + j*3+1, a*s1);
    atomicAdd(agv + j*3+2, a*s2);
  }
}

// ------------------------------------ o3-linear, gate, residual, BN statistics
__global__ void __launch_bounds__(32) node_kernel(
    const float* se, const float* aggs, const float* aggv, const float* xsf,
    const float* oWs, const float* oWv, const float* gWs, const float* gWv,
    float* sbuf, float* vbuf, float* bnsum, int N)
{
  __shared__ float as[HS], av[48], os[HS], ov[48], gs[48], gv[48];
  int i = blockIdx.x;
  int j = threadIdx.x;
  float inv = fast_rcp(se[i] + 1e-8f);       // agg = (sum ex*msg)/(sum ex + eps)
  as[j] = aggs[(size_t)i*HS + j]*inv;
  if (j < HV){
    av[j*3+0] = aggv[(size_t)i*48 + j*3+0]*inv;
    av[j*3+1] = aggv[(size_t)i*48 + j*3+1]*inv;
    av[j*3+2] = aggv[(size_t)i*48 + j*3+2]*inv;
  }
  __syncthreads();
  { float acc=0; for (int k=0;k<HS;k++) acc += as[k]*oWs[k*HS+j]; os[j]=acc*INV_SQRT_HS; }
  if (j < HV){
    for (int d=0;d<3;d++){
      float acc=0; for (int k=0;k<HV;k++) acc += av[k*3+d]*oWv[k*HV+j];
      ov[j*3+d] = acc*INV_SQRT_HV;
    }
  }
  __syncthreads();
  for (int c=j; c<48; c+=32){
    float acc=0; for (int k=0;k<HS;k++) acc += os[k]*gWs[k*48+c];
    gs[c] = acc*INV_SQRT_HS;
  }
  if (j < HV){
    for (int d=0;d<3;d++){
      float acc=0; for (int k=0;k<HV;k++) acc += ov[k*3+d]*gWv[k*HV+j];
      gv[j*3+d] = acc*INV_SQRT_HV;
    }
  }
  __syncthreads();
  float s = silu_f(gs[j]) + xsf[(size_t)i*HS + j];   // residual (x_v == 0)
  sbuf[(size_t)i*HS + j] = s;
  atomicAdd(bnsum + j, s);
  atomicAdd(bnsum + HS + j, s*s);
  if (j < HV){
    float g = sigmoid_f(gs[HS+j]);
    float n2 = 0.0f;
    for (int d=0;d<3;d++){
      float vv = g*gv[j*3+d];
      vbuf[(size_t)i*48 + j*3+d] = vv;
      n2 += vv*vv;
    }
    atomicAdd(bnsum + 64 + j, n2);
  }
}

__global__ void bn_stats_kernel(const float* bnsum, const float* bn_ws, const float* bn_wv,
                                float* stats, int N){
  int t = threadIdx.x;
  float invN = fast_rcp((float)N);
  if (t < HS){
    float mu  = bnsum[t]*invN;
    float var = bnsum[HS+t]*invN - mu*mu;
    stats[t]      = mu;
    stats[HS+t]   = bn_ws[t]*rsqrtf(var + 1e-5f);
  } else if (t < HS+HV){
    int j = t - HS;
    float n2 = bnsum[64+j]*invN;
    stats[64+j] = (bn_wv[j]*bn_wv[j])*fast_rcp(n2 + 1e-5f);  // scale^2 for vector norms
  }
}

__global__ void pool_kernel(const float* sbuf, const float* vbuf, const float* stats,
                            const float* bn_bs, const int* batch_idx,
                            float* pooled, float* cnt, int N){
  int i = blockIdx.x*blockDim.x + threadIdx.x;
  if (i >= N) return;
  int b = batch_idx[i];
  float* pp = pooled + (size_t)b*48;
  for (int j=0;j<HS;j++){
    float sn = (sbuf[(size_t)i*HS+j] - stats[j])*stats[HS+j] + bn_bs[j];
    atomicAdd(pp + j, sn);
  }
  for (int j=0;j<HV;j++){
    float r2 = 0.0f;
    for (int d=0;d<3;d++){ float vv = vbuf[(size_t)i*48 + j*3+d]; r2 += vv*vv; }
    atomicAdd(pp + HS + j, sqrtf(r2*stats[64+j] + 1e-12f));
  }
  atomicAdd(cnt + b, 1.0f);
}

__global__ void __launch_bounds__(256) head_kernel(
    const float* pooled, const float* cnt,
    const float* h1W, const float* h1b,
    const float* h2W, const float* h2b,
    const float* h3W, const float* h3b,
    float* out, int B)
{
  __shared__ float p[16*48];
  __shared__ float hb1[16*256];
  __shared__ float hb2[16*128];
  int t = threadIdx.x;
  for (int idx=t; idx<B*48; idx+=256){
    int r = idx/48;
    p[idx] = pooled[idx]*fast_rcp(fmaxf(cnt[r], 1.0f));
  }
  __syncthreads();
  for (int idx=t; idx<B*256; idx+=256){
    int r = idx>>8, c = idx&255;
    float acc = h1b[c];
    for (int k=0;k<48;k++) acc += p[r*48+k]*h1W[k*256+c];
    hb1[idx] = silu_f(acc);
  }
  __syncthreads();
  for (int idx=t; idx<B*128; idx+=256){
    int r = idx>>7, c = idx&127;
    float acc = h2b[c];
    for (int k=0;k<256;k++) acc += hb1[r*256+k]*h2W[k*128+c];
    hb2[idx] = silu_f(acc);
  }
  __syncthreads();
  for (int idx=t; idx<B*18; idx+=256){
    int r = idx/18, c = idx%18;
    float acc = h3b[c];
    for (int k=0;k<128;k++) acc += hb2[r*128+k]*h3W[k*18+c];
    out[idx] = acc;
  }
}

// ------------------------------------------------------------------ launcher
extern "C" void kernel_launch(void* const* d_in, const int* in_sizes, int n_in,
                              void* d_out, int out_size, void* d_ws, size_t ws_size,
                              hipStream_t stream) {
  const float* features = (const float*)d_in[0];
  const float* positions= (const float*)d_in[1];
  const float* embed_W  = (const float*)d_in[2];
  const float* embed_b  = (const float*)d_in[3];
  const float* r1_W     = (const float*)d_in[4];
  const float* r1_b     = (const float*)d_in[5];
  const float* r2_W     = (const float*)d_in[6];
  const float* r2_b     = (const float*)d_in[7];
  const float* q_W      = (const float*)d_in[8];
  const float* q_b      = (const float*)d_in[9];
  const float* k_W      = (const float*)d_in[10];
  const float* k_b      = (const float*)d_in[11];
  const float* o_Ws     = (const float*)d_in[12];
  const float* o_Wv     = (const float*)d_in[13];
  const float* g_Ws     = (const float*)d_in[14];
  const float* g_Wv     = (const float*)d_in[15];
  const float* bn_ws    = (const float*)d_in[16];
  const float* bn_bs    = (const float*)d_in[17];
  const float* bn_wv    = (const float*)d_in[18];
  const float* h1_W     = (const float*)d_in[19];
  const float* h1_b     = (const float*)d_in[20];
  const float* h2_W     = (const float*)d_in[21];
  const float* h2_b     = (const float*)d_in[22];
  const float* h3_W     = (const float*)d_in[23];
  const float* h3_b     = (const float*)d_in[24];
  const int*   ei       = (const int*)  d_in[25];
  const int*   batch_idx= (const int*)  d_in[26];
  float* out = (float*)d_out;

  const int N = in_sizes[0] / 46;
  const int E = in_sizes[25] / 2;
  const int B = out_size / 18;

  char* base = (char*)d_ws;
  size_t off = 0;
  auto alloc = [&](size_t bytes)->void*{
    void* p = base + off;
    off = (off + bytes + 255) & ~(size_t)255;
    return p;
  };

  bf16*  R      = (bf16*) alloc((size_t)KTOT*HS*RSTRIDE*sizeof(bf16));
  bf16*  r1Wb   = (bf16*) alloc((size_t)HS*128*sizeof(bf16));
  bf16*  xsb    = (bf16*) alloc((size_t)N*HS*sizeof(bf16));
  float* xsf    = (float*)alloc((size_t)N*HS*sizeof(float));
  float* nodeq  = (float*)alloc((size_t)N*NH*sizeof(float));
  float* msg_s  = (float*)alloc((size_t)E*HS*sizeof(float));   // reused as node sbuf
  float* a_sv   = (float*)alloc((size_t)E*HV*sizeof(float));   // reused as node vbuf
  float* shv    = (float*)alloc((size_t)E*4*sizeof(float));
  float* logit  = (float*)alloc((size_t)E*sizeof(float));
  float* stats  = (float*)alloc(80*sizeof(float));
  // contiguous zero-initialized block
  size_t zfloats = (size_t)N + N + 32ull*N + 48ull*N + 80 + 48ull*B + B;
  float* zbase  = (float*)alloc(zfloats*sizeof(float));
  unsigned* nmax = (unsigned*)zbase;
  float* se     = zbase + N;
  float* aggs   = se + N;
  float* aggv   = aggs + 32ull*N;
  float* bnsum  = aggv + 48ull*N;
  float* pooled = bnsum + 80;
  float* cnt    = pooled + 48ull*B;

  zero_kernel<<<(int)((zfloats+255)/256), 256, 0, stream>>>(zbase, (long)zfloats);
  prep_r1_kernel<<<(HS*128+255)/256, 256, 0, stream>>>(r1_W, r1Wb);
  {
    int total = KTOT*HS*RSTRIDE;
    prep_R_kernel<<<(total+255)/256, 256, 0, stream>>>(r2_W, r2_b, R);
  }
  embed_kernel<<<(N+63)/64, 64, 0, stream>>>(features, embed_W, embed_b, q_W, q_b,
                                             xsf, xsb, nodeq, N);
  {
    int ngroups = (E + 31)/32;
    int blocks  = (ngroups + WPB - 1)/WPB;
    edge_kernel<<<blocks, 64, 0, stream>>>(ei, positions, xsb, nodeq, r1Wb, r1_b, R,
                                           k_W, k_b, msg_s, a_sv, shv, logit,
                                           nmax, E, ngroups);
  }
  softmax_acc_kernel<<<(E+255)/256, 256, 0, stream>>>(ei, logit, nmax, msg_s, a_sv, shv,
                                                      se, aggs, aggv, E);
  // per-edge buffers are dead now: reuse msg_s as node sbuf (N*32 <= E*32),
  // a_sv as node vbuf (N*48 <= E*16 since E >= 3N).
  node_kernel<<<N, 32, 0, stream>>>(se, aggs, aggv, xsf, o_Ws, o_Wv, g_Ws, g_Wv,
                                    msg_s, a_sv, bnsum, N);
  bn_stats_kernel<<<1, 64, 0, stream>>>(bnsum, bn_ws, bn_wv, stats, N);
  pool_kernel<<<(N+255)/256, 256, 0, stream>>>(msg_s, a_sv, stats, bn_bs,
                                               batch_idx, pooled, cnt, N);
  head_kernel<<<1, 256, 0, stream>>>(pooled, cnt, h1_W, h1_b, h2_W, h2_b, h3_W, h3_b,
                                     out, B);
}
